// _PixelCNN_63170378989964
// MI455X (gfx1250) — matrix operand order repaired and checked by reading, not compiled, not run on hardware
//
#include <hip/hip_runtime.h>

typedef __attribute__((ext_vector_type(16))) _Float16 v16h;
typedef __attribute__((ext_vector_type(8)))  _Float16 v8h;
typedef __attribute__((ext_vector_type(8)))  float    v8f;
typedef __attribute__((ext_vector_type(4)))  int      v4i;

#define BATCH 64
#define HGT   128
#define WID   128
#define HID   64
#define CONDC 16
#define NB    4

// ---------------------------------------------------------------------------
// Pack masked 3x3 weights (mask 'B': taps (0,0)(0,1)(0,2)(1,0)(1,1)) into the
// contiguous-K B-fragment order used with V_WMMA_F32_16X16X32_F16 (same as A):
//   lane l<16 : b[j] = B[K=j   ][N=l   ]   (j=0..15)
//   lane l>=16: b[j] = B[K=16+j][N=l-16]
// K-chunk k (0..9): tap t = k/2, channel base = (k&1)*32.
// bpack layout: [layer][k][ntile][lane][16] halves.
// ---------------------------------------------------------------------------
__global__ __launch_bounds__(256) void pack_weights(
    const float* __restrict__ wB, _Float16* __restrict__ bpack)
{
    int idx = blockIdx.x * 256 + threadIdx.x;
    if (idx >= NB * 10 * 4 * 32 * 16) return;
    int j    = idx & 15;
    int l    = (idx >> 4) & 31;
    int nt   = (idx >> 9) & 3;
    int lk   = idx >> 11;          // layer*10 + k
    int layer = lk / 10;
    int k     = lk - layer * 10;
    int t  = k >> 1;
    int cc = k & 1;
    int kh = (t < 3) ? 0 : 1;
    int kw = (t < 3) ? t : (t - 3);
    int co = nt * 16 + (l & 15);
    int ci = cc * 32 + ((l >> 4) << 4) + j;
    float v = wB[((((size_t)layer * HID + co) * HID + ci) * 3 + kh) * 3 + kw];
    bpack[idx] = (_Float16)v;
}

// ---------------------------------------------------------------------------
// Layer 0: 7x7 mask-A conv (1 ch, 24 live taps) + cond 1x1 (16->64) + ReLU.
// Output f16 NHWC [B,H,W,64]. Weights staged in LDS.
// ---------------------------------------------------------------------------
__global__ __launch_bounds__(256) void layer0(
    const float* __restrict__ samples, const float* __restrict__ params,
    const float* __restrict__ w0, const float* __restrict__ b0,
    const float* __restrict__ wc, const float* __restrict__ bc,
    _Float16* __restrict__ hout)
{
    __shared__ float sw[24 * 64];
    __shared__ float swc[16 * 64];
    __shared__ float sb[64];

    for (int i = threadIdx.x; i < 24 * 64; i += 256) {
        int t  = i >> 6, co = i & 63;
        int kh = (t < 21) ? (t / 7) : 3;
        int kw = (t < 21) ? (t - kh * 7) : (t - 21);
        sw[i] = w0[co * 49 + kh * 7 + kw];
    }
    for (int i = threadIdx.x; i < 16 * 64; i += 256) {
        int ci = i >> 6, co = i & 63;
        swc[i] = wc[co * 16 + ci];
    }
    if (threadIdx.x < 64) sb[threadIdx.x] = b0[threadIdx.x] + bc[threadIdx.x];
    __syncthreads();

    int p  = blockIdx.x * 256 + threadIdx.x;   // pixel id
    int b  = p >> 14;
    int y  = (p >> 7) & 127;
    int xo = p & 127;
    const float* simg = samples + ((size_t)b << 14);

    float acc[64];
#pragma unroll
    for (int co = 0; co < 64; ++co) acc[co] = sb[co];

    for (int t = 0; t < 24; ++t) {
        int kh = (t < 21) ? (t / 7) : 3;
        int kw = (t < 21) ? (t - kh * 7) : (t - 21);
        int yy = y + kh - 3, xx = xo + kw - 3;
        float xv = 0.f;
        if (yy >= 0 && xx >= 0 && xx < WID)
            xv = simg[yy * WID + xx] * 2.f - 1.f;   // {0,1} -> {-1,1}, pad = 0
#pragma unroll
        for (int co = 0; co < 64; ++co) acc[co] += sw[t * 64 + co] * xv;
    }
    for (int ci = 0; ci < CONDC; ++ci) {
        float pv = params[(((size_t)b * CONDC + ci) << 14) + (y << 7) + xo];
#pragma unroll
        for (int co = 0; co < 64; ++co) acc[co] += swc[ci * 64 + co] * pv;
    }

    _Float16* ob = hout + ((size_t)p << 6);
#pragma unroll
    for (int co = 0; co < 64; ++co) {
        float v = acc[co] > 0.f ? acc[co] : 0.f;
        ob[co] = (_Float16)v;
    }
}

// ---------------------------------------------------------------------------
// Masked-B 3x3 conv 64->64 + ReLU as implicit GEMM on v_wmma_f32_16x16x32_f16.
// Block = 256 threads = 8 waves = 2 image rows; each wave owns a 32-pixel
// x-tile (two M=16 WMMA tiles sharing every B fragment).
// B fragments (40 KB) are staged to LDS once per block — async global->LDS
// path when the toolchain exposes it, plain copy otherwise.
// ---------------------------------------------------------------------------
__global__ __launch_bounds__(256) void conv3x3_wmma(
    const _Float16* __restrict__ hin, _Float16* __restrict__ hout,
    const _Float16* __restrict__ bpack,   // this layer: [10][4][32][16] halves
    const float* __restrict__ bias)       // [64]
{
    __shared__ __attribute__((aligned(64))) _Float16 sB[10 * 4 * 32 * 16]; // 40 KB

    const int tid = threadIdx.x;

#if defined(__gfx1250__) && __has_builtin(__builtin_amdgcn_global_load_async_to_lds_b128) && __has_builtin(__builtin_amdgcn_s_wait_asynccnt)
    {
        typedef __attribute__((address_space(1))) v4i* gv4i_p;
        typedef __attribute__((address_space(3))) v4i* lv4i_p;
        for (int i = tid; i < 2560; i += 256) {
            gv4i_p g = (gv4i_p)(void*)(bpack + (size_t)i * 8);   // i*16 bytes
            lv4i_p l = (lv4i_p)(sB + (size_t)i * 8);
            __builtin_amdgcn_global_load_async_to_lds_b128(g, l, 0, 0);
        }
        __builtin_amdgcn_s_wait_asynccnt(0);
    }
#else
    for (int i = tid; i < 2560; i += 256)
        ((float4*)sB)[i] = ((const float4*)bpack)[i];
#endif
    __syncthreads();

    const int lane = tid & 31;
    const int wave = tid >> 5;
    const int m    = lane & 15;           // A-row / C-column group index
    const int hi   = lane >> 4;           // half-wave selector
    const int row  = wave >> 2;           // 0..1 within block
    const int by   = blockIdx.x * 2 + row;  // b*H + y
    const int y    = by & (HGT - 1);
    const int x0   = (wave & 3) << 5;     // 32-pixel tile base

    // next-row prefetch (gfx1250 global_prefetch_b8)
    int pby = (by + 2 < BATCH * HGT) ? (by + 2) : by;
    __builtin_prefetch(hin + (((size_t)pby * WID + x0 + m) << 6), 0, 1);

    v8f acc[2][4];
#pragma unroll
    for (int nt = 0; nt < 4; ++nt) {
        float bv = bias[nt * 16 + m];
#pragma unroll
        for (int ti = 0; ti < 2; ++ti)
#pragma unroll
            for (int r = 0; r < 8; ++r) acc[ti][nt][r] = bv;
    }

    const int DY[5] = {-1, -1, -1, 0, 0};
    const int DX[5] = {-1,  0,  1, -1, 0};

#pragma unroll
    for (int t = 0; t < 5; ++t) {
        const int yy = y + DY[t];
        const _Float16* abase[2];
        _Float16 mv[2];
#pragma unroll
        for (int ti = 0; ti < 2; ++ti) {
            int xb = x0 + ti * 16 + m;     // A-row M=m of tile ti -> pixel xb
            int xx = xb + DX[t];
            bool valid = (yy >= 0) && (xx >= 0) && (xx < WID);
            // branchless: clamp to a safe in-bounds pixel, zero the frag below
            int pix = valid ? ((by + DY[t]) * WID + xx) : (by * WID + xb);
            abase[ti] = hin + ((size_t)pix << 6);
            mv[ti] = (_Float16)(valid ? 1.f : 0.f);
        }

#pragma unroll
        for (int cc = 0; cc < 2; ++cc) {
            const int k = t * 2 + cc;
            // A fragment (contiguous K, matching bpack):
            //  lane<16 : a[0..15]=K0..15   (K == cin-in-chunk)
            //  lane>=16: a[0..15]=K16..31
            v16h a[2];
#pragma unroll
            for (int ti = 0; ti < 2; ++ti) {
                const v8h* ap = (const v8h*)(abase[ti] + cc * 32 + hi * 16);
                v8h alo = ap[0];
                v8h ahi = ap[1];          // +8 halves
#pragma unroll
                for (int j = 0; j < 8; ++j) {
                    a[ti][j]     = alo[j] * mv[ti];
                    a[ti][8 + j] = ahi[j] * mv[ti];
                }
            }
#pragma unroll
            for (int nt = 0; nt < 4; ++nt) {
                v16h bf = *(const v16h*)(sB + (((k * 4 + nt) * 32 + lane) << 4));
                acc[0][nt] = __builtin_amdgcn_wmma_f32_16x16x32_f16(
                    false, a[0], false, bf, (short)0, acc[0][nt], false, false);
                acc[1][nt] = __builtin_amdgcn_wmma_f32_16x16x32_f16(
                    false, a[1], false, bf, (short)0, acc[1][nt], false, false);
            }
        }
    }

    // ReLU + f16 store. C/D layout: VGPR r, lane<16 -> C[M=r][N=lane];
    // lane>=16 -> C[M=8+r][N=lane-16].
#pragma unroll
    for (int ti = 0; ti < 2; ++ti) {
#pragma unroll
        for (int nt = 0; nt < 4; ++nt) {
            int co = nt * 16 + m;
#pragma unroll
            for (int r = 0; r < 8; ++r) {
                int px = x0 + ti * 16 + r + hi * 8;
                float v = acc[ti][nt][r];
                v = v > 0.f ? v : 0.f;
                hout[(((size_t)by) * WID + px) * 64 + co] = (_Float16)v;
            }
        }
    }
}

// ---------------------------------------------------------------------------
// Output head: 1x1 conv 64->1, Bernoulli log-prob lp = s*L - softplus(L),
// deterministic block reduction over one image row -> partial[b*H + y].
// ---------------------------------------------------------------------------
__global__ __launch_bounds__(128) void out_lp(
    const _Float16* __restrict__ h, const float* __restrict__ w_out,
    const float* __restrict__ b_out, const float* __restrict__ samples,
    float* __restrict__ partial)
{
    __shared__ float wsh[64];
    __shared__ float red[128];
    int tid = threadIdx.x;
    if (tid < 64) wsh[tid] = w_out[tid];
    __syncthreads();

    int by = blockIdx.x;                  // b*H + y
    const v8h* hp = (const v8h*)(h + (((size_t)by * WID + tid) << 6));
    float L = b_out[0];
#pragma unroll
    for (int c8 = 0; c8 < 8; ++c8) {
        v8h hv = hp[c8];
#pragma unroll
        for (int j = 0; j < 8; ++j) L += (float)hv[j] * wsh[c8 * 8 + j];
    }
    float s  = samples[((size_t)by << 7) + tid];
    float sp = fmaxf(L, 0.f) + log1pf(expf(-fabsf(L)));   // softplus(L)
    red[tid] = s * L - sp;
    __syncthreads();
#pragma unroll
    for (int off = 64; off > 0; off >>= 1) {
        if (tid < off) red[tid] += red[tid + off];
        __syncthreads();
    }
    if (tid == 0) partial[by] = red[0];
}

__global__ __launch_bounds__(128) void final_red(
    const float* __restrict__ partial, float* __restrict__ out)
{
    __shared__ float red[128];
    int tid = threadIdx.x;
    red[tid] = partial[(blockIdx.x << 7) + tid];
    __syncthreads();
#pragma unroll
    for (int off = 64; off > 0; off >>= 1) {
        if (tid < off) red[tid] += red[tid + off];
        __syncthreads();
    }
    if (tid == 0) out[blockIdx.x] = red[0];
}

// ---------------------------------------------------------------------------
extern "C" void kernel_launch(void* const* d_in, const int* in_sizes, int n_in,
                              void* d_out, int out_size, void* d_ws, size_t ws_size,
                              hipStream_t stream)
{
    const float* samples = (const float*)d_in[0];
    const float* params  = (const float*)d_in[1];
    const float* w0      = (const float*)d_in[2];
    const float* b0      = (const float*)d_in[3];
    const float* wc      = (const float*)d_in[4];
    const float* bc      = (const float*)d_in[5];
    const float* wB      = (const float*)d_in[6];
    const float* bB      = (const float*)d_in[7];
    const float* w_out   = (const float*)d_in[8];
    const float* b_out   = (const float*)d_in[9];
    float* out = (float*)d_out;

    const size_t HSZ     = (size_t)BATCH * HGT * WID * HID * sizeof(_Float16); // 134 MB
    const size_t PACKSZ  = (size_t)NB * 10 * 4 * 32 * 16 * sizeof(_Float16);   // 160 KB
    const size_t PARTSZ  = (size_t)BATCH * HGT * sizeof(float);                // 32 KB
    if (ws_size < 2 * HSZ + PACKSZ + PARTSZ) return;  // workspace too small

    char* ws = (char*)d_ws;
    _Float16* hA    = (_Float16*)ws;
    _Float16* hB    = (_Float16*)(ws + HSZ);
    _Float16* bpack = (_Float16*)(ws + 2 * HSZ);
    float* partial  = (float*)(ws + 2 * HSZ + PACKSZ);

    pack_weights<<<(NB * 10 * 4 * 32 * 16 + 255) / 256, 256, 0, stream>>>(wB, bpack);
    layer0<<<(BATCH * HGT * WID) / 256, 256, 0, stream>>>(samples, params, w0, b0, wc, bc, hA);

    _Float16* cur = hA;
    _Float16* nxt = hB;
    for (int i = 0; i < NB; ++i) {
        conv3x3_wmma<<<(BATCH * HGT) / 2, 256, 0, stream>>>(
            cur, nxt, bpack + (size_t)i * 10 * 4 * 32 * 16, bB + i * HID);
        _Float16* t = cur; cur = nxt; nxt = t;
    }

    out_lp<<<BATCH * HGT, 128, 0, stream>>>(cur, w_out, b_out, samples, partial);
    final_red<<<BATCH, 128, 0, stream>>>(partial, out);
}
